// PVCNN2Base_4475355922676
// MI455X (gfx1250) — matrix operand order repaired and checked by reading, not compiled, not run on hardware
//
#include <hip/hip_runtime.h>
#include <cstdint>
#include <cstddef>

// ---------------- CDNA5 types ----------------
typedef __attribute__((ext_vector_type(16))) _Float16 v16h;
typedef __attribute__((ext_vector_type(8)))  _Float16 v8h;
typedef __attribute__((ext_vector_type(8)))  float    v8f;
typedef unsigned int u32x4 __attribute__((ext_vector_type(4)));
typedef int          s32x4 __attribute__((ext_vector_type(4)));
typedef int          s32x8 __attribute__((ext_vector_type(8)));

#if defined(__has_builtin)
#if __has_builtin(__builtin_amdgcn_tensor_load_to_lds) && __has_builtin(__builtin_amdgcn_s_wait_tensorcnt)
#define PV_TDM 1
#endif
#endif

#define GEMM_KMAX 832  // largest padded C_in (FP stage 0 mlp_in = 832, already 32-aligned)

#define SHUF16(lo, hi) \
  __builtin_shufflevector(lo, hi, 0, 1, 2, 3, 4, 5, 6, 7, 8, 9, 10, 11, 12, 13, 14, 15)

// ---------------- staging kernels (f32 -> padded/transposed f16) ----------------
// Wh[o, k] (Op x Kp, zero padded)
__global__ void k_wpad(_Float16* __restrict__ Wh, const float* __restrict__ W,
                       int O, int K, int Op, int Kp) {
  int t = blockIdx.x * 256 + threadIdx.x;
  if (t >= Op * Kp) return;
  int k = t % Kp, o = t / Kp;
  Wh[t] = (o < O && k < K) ? (_Float16)W[(size_t)o * K + k] : (_Float16)0;
}
// Xt[b, n, k] = X[b, k, n]  (Nnp x Kp, zero padded)
__global__ void k_xtr(_Float16* __restrict__ Xt, const float* __restrict__ X,
                      int K, int Kp, int Nn, int Nnp, int B_) {
  int t = blockIdx.x * 256 + threadIdx.x;
  if (t >= B_ * Nnp * Kp) return;
  int k = t % Kp; int n = (t / Kp) % Nnp; int b = t / (Kp * Nnp);
  Xt[t] = (k < K && n < Nn) ? (_Float16)X[((size_t)b * K + k) * Nn + n] : (_Float16)0;
}
// Wt[o, tap, c] = W[o, c, tap]  (Op x 27 x Cp, zero padded)
__global__ void k_wconv(_Float16* __restrict__ Wt, const float* __restrict__ W,
                        int O, int Cin, int Op, int Cp) {
  int t = blockIdx.x * 256 + threadIdx.x;
  if (t >= Op * 27 * Cp) return;
  int c = t % Cp; int tap = (t / Cp) % 27; int o = t / (Cp * 27);
  Wt[t] = (o < O && c < Cin) ? (_Float16)W[((size_t)o * Cin + c) * 27 + tap] : (_Float16)0;
}
// Gt[b, v, c] = G[b, c, v]  (r3 x Cp, zero padded)
__global__ void k_gtr(_Float16* __restrict__ Gt, const float* __restrict__ G,
                      int Cin, int Cp, int r3, int B_) {
  int t = blockIdx.x * 256 + threadIdx.x;
  if (t >= B_ * r3 * Cp) return;
  int c = t % Cp; int v = (t / Cp) % r3; int b = t / (Cp * r3);
  Gt[t] = (c < Cin) ? (_Float16)G[((size_t)b * Cin + c) * r3 + v] : (_Float16)0;
}

// ---------------- small utility kernels ----------------
__global__ void k_fill(float* p, float v, int n) {
  int t = blockIdx.x * 256 + threadIdx.x;
  if (t < n) p[t] = v;
}
__global__ void k_add(float* a, const float* b, int n) {
  int t = blockIdx.x * 256 + threadIdx.x;
  if (t < n) a[t] += b[t];
}
__global__ void k_copy_ch(float* __restrict__ dst, const float* __restrict__ src,
                          int off, int Cd, int Cs, int S, int B_) {
  int t = blockIdx.x * 256 + threadIdx.x;
  if (t >= B_ * Cs * S) return;
  int s = t % S; int c = (t / S) % Cs; int b = t / (S * Cs);
  dst[((size_t)b * Cd + off + c) * S + s] = src[((size_t)b * Cs + c) * S + s];
}
__global__ void k_bcast(float* __restrict__ dst, const float* __restrict__ te,
                        int off, int Cd, int S, int B_) {
  int t = blockIdx.x * 256 + threadIdx.x;
  if (t >= B_ * 64 * S) return;
  int s = t % S; int e = (t / S) % 64; int b = t / (S * 64);
  dst[((size_t)b * Cd + off + e) * S + s] = te[b * 64 + e];
}

// ---------------- WMMA GEMM:  Y[b,o,n] = sum_k W[o,k]*X[b,k,n] + bias[o] ----------------
// One wave32 per 16(O) x 64(N) tile: 4 independent accumulators, A reused 4x.
// A: [16,Kp] weight tile in LDS (staged by the Tensor Data Mover), two ds_load_b128 per k-step.
// B: activations transposed [B,Nnp,Kp]; all 8 b128 loads issued before the 4 WMMAs.
__global__ void k_gemm_wmma(const _Float16* __restrict__ Wh,  // [Op,Kp]
                            const _Float16* __restrict__ Xt,  // [B,Nnp,Kp]
                            const float* __restrict__ bias,   // [O] or null
                            float* __restrict__ Y,            // [B,O,Nn]
                            int O, int Kp, int Nn, int Nnp) {
  __shared__ __align__(16) _Float16 Wl[16 * GEMM_KMAX];
  int lane = threadIdx.x & 31;
  int g = lane >> 4, mn = lane & 15;
  int n0 = blockIdx.x * 64, o0 = blockIdx.y * 16, bb = blockIdx.z;
#ifdef PV_TDM
  {
    // TDM descriptor: 2D tile, 16 rows x Kp halves, row stride Kp (source fully padded)
    unsigned long long ga = (unsigned long long)(uintptr_t)(Wh + (size_t)o0 * Kp);
    unsigned lds = (unsigned)(uintptr_t)&Wl[0];
    u32x4 g0;
    g0[0] = 1u;                                   // count=1 valid descriptor
    g0[1] = lds;                                  // lds_addr
    g0[2] = (unsigned)ga;                         // global_addr[31:0]
    g0[3] = (unsigned)(((ga >> 32) & 0x1FFFFFFull) | (2u << 30));  // addr[56:32] | type=2
    unsigned dim0 = (unsigned)Kp;
    unsigned dim1 = 16u;
    s32x8 g1;
    g1[0] = (int)(1u << 16);                                            // data_size=1 (2B)
    g1[1] = (int)((dim0 & 0xFFFFu) << 16);                              // tensor_dim0 lo16
    g1[2] = (int)(((dim0 >> 16) & 0xFFFFu) | ((dim1 & 0xFFFFu) << 16));
    g1[3] = (int)(((dim1 >> 16) & 0xFFFFu) | ((dim0 & 0xFFFFu) << 16)); // tile_dim0=Kp
    g1[4] = 16;                                                         // tile_dim1=16
    g1[5] = (int)dim0;                                                  // dim0_stride=Kp
    g1[6] = 0; g1[7] = 0;
    s32x4 g2; g2[0] = g2[1] = g2[2] = g2[3] = 0;
    s32x4 g3; g3[0] = g3[1] = g3[2] = g3[3] = 0;
#if __clang_major__ >= 23
    s32x8 g4; g4[0]=g4[1]=g4[2]=g4[3]=g4[4]=g4[5]=g4[6]=g4[7]=0;
    __builtin_amdgcn_tensor_load_to_lds(g0, g1, g2, g3, g4, 0);
#else
    __builtin_amdgcn_tensor_load_to_lds(g0, g1, g2, g3, 0);
#endif
    __builtin_amdgcn_s_wait_tensorcnt((short)0);
  }
#else
  for (int t = threadIdx.x * 8; t < 16 * Kp; t += 32 * 8)
    *(v8h*)&Wl[t] = *(const v8h*)&Wh[(size_t)o0 * Kp + t];
#endif
  __syncthreads();
  const _Float16* Xb = Xt + (size_t)bb * Nnp * Kp;
  const _Float16* wrow = &Wl[mn * Kp];
  v8f acc[4] = {{}, {}, {}, {}};
  for (int k0 = 0; k0 < Kp; k0 += 32) {
    v8h alo = *(const v8h*)(wrow + k0 + g * 8);
    v8h ahi = *(const v8h*)(wrow + k0 + 16 + g * 8);
    v16h a = SHUF16(alo, ahi);
    if (k0 + 32 < Kp)
      __builtin_prefetch(Xb + (size_t)(n0 + mn) * Kp + k0 + 32, 0, 1);
    v16h bvs[4];
#pragma unroll
    for (int t = 0; t < 4; t++) {
      const _Float16* xp = Xb + (size_t)(n0 + t * 16 + mn) * Kp + k0 + g * 8;
      v8h blo = *(const v8h*)xp;
      v8h bhi = *(const v8h*)(xp + 16);
      bvs[t] = SHUF16(blo, bhi);
    }
#pragma unroll
    for (int t = 0; t < 4; t++)
      acc[t] = __builtin_amdgcn_wmma_f32_16x16x32_f16(false, a, false, bvs[t], (short)0,
                                                      acc[t], false, false);
  }
#pragma unroll
  for (int t = 0; t < 4; t++) {
    int n = n0 + t * 16 + mn;
    if (n >= Nn) continue;
#pragma unroll
    for (int r = 0; r < 8; r++) {
      int m = o0 + r + g * 8;
      if (m < O) Y[((size_t)bb * O + m) * Nn + n] = acc[t][r] + (bias ? bias[m] : 0.f);
    }
  }
}

// ---------------- WMMA 3x3x3 conv ('SAME') over voxel grid ----------------
// Weights [Op,27,Cp] tap-major; grid transposed [B,r3,Cp] so B-fragments are contiguous.
__global__ void k_conv3d_wmma(const _Float16* __restrict__ Wt,
                              const _Float16* __restrict__ Gt,
                              const float* __restrict__ bias,
                              float* __restrict__ Y,  // [B,O,r3]
                              int O, int Cp, int r) {
  int lane = threadIdx.x & 31;
  int g = lane >> 4, mn = lane & 15;
  int r2 = r * r, r3 = r2 * r;
  int n0 = blockIdx.x * 64, o0 = blockIdx.y * 16, bb = blockIdx.z;
  const _Float16* Gb = Gt + (size_t)bb * r3 * Cp;
  int vxs[4], vys[4], vzs[4];
#pragma unroll
  for (int t = 0; t < 4; t++) {
    int n = n0 + t * 16 + mn;
    vzs[t] = n % r; vys[t] = (n / r) % r; vxs[t] = n / r2;
  }
  v8f acc[4] = {{}, {}, {}, {}};
  const _Float16* wbase = Wt + (size_t)(o0 + mn) * 27 * Cp;
  for (int tap = 0; tap < 27; tap++) {
    int dx = tap / 9 - 1, dy = (tap / 3) % 3 - 1, dz = tap % 3 - 1;
    int sof[4]; bool inb[4];
#pragma unroll
    for (int t = 0; t < 4; t++) {
      int sx = vxs[t] + dx, sy = vys[t] + dy, sz = vzs[t] + dz;
      inb[t] = sx >= 0 && sx < r && sy >= 0 && sy < r && sz >= 0 && sz < r;
      sof[t] = (sx * r + sy) * r + sz;
    }
    const _Float16* wt = wbase + (size_t)tap * Cp;
    for (int c0 = 0; c0 < Cp; c0 += 32) {
      v8h alo = *(const v8h*)(wt + c0 + g * 8);
      v8h ahi = *(const v8h*)(wt + c0 + 16 + g * 8);
      v16h a = SHUF16(alo, ahi);
      v16h bvs[4];
#pragma unroll
      for (int t = 0; t < 4; t++) {
        v8h blo = {}, bhi = {};
        if (inb[t]) {
          const _Float16* gp = Gb + (size_t)sof[t] * Cp + c0 + g * 8;
          blo = *(const v8h*)gp;
          bhi = *(const v8h*)(gp + 16);
        }
        bvs[t] = SHUF16(blo, bhi);
      }
#pragma unroll
      for (int t = 0; t < 4; t++)
        acc[t] = __builtin_amdgcn_wmma_f32_16x16x32_f16(false, a, false, bvs[t], (short)0,
                                                        acc[t], false, false);
    }
  }
#pragma unroll
  for (int t = 0; t < 4; t++) {
    int n = n0 + t * 16 + mn;
#pragma unroll
    for (int rr = 0; rr < 8; rr++) {
      int m = o0 + rr + g * 8;
      if (m < O) Y[((size_t)bb * O + m) * r3 + n] = acc[t][rr] + (bias ? bias[m] : 0.f);
    }
  }
}

// ---------------- GroupNorm (G=8) + swish, in place, x: [B,C,S] ----------------
__global__ void k_groupnorm(float* __restrict__ x, const float* __restrict__ gam,
                            const float* __restrict__ bet, int C, int S, int dosw) {
  __shared__ float r1[256], r2[256];
  __shared__ float m_, v_;
  int gI = blockIdx.x, b = blockIdx.y, tid = threadIdx.x;
  int cg = C / 8;
  size_t base = ((size_t)b * C + (size_t)gI * cg) * S;
  long long len = (long long)cg * S;
  float s1 = 0.f, s2 = 0.f;
  for (long long i = tid; i < len; i += 256) { float v = x[base + i]; s1 += v; s2 += v * v; }
  r1[tid] = s1; r2[tid] = s2; __syncthreads();
  for (int s = 128; s > 0; s >>= 1) {
    if (tid < s) { r1[tid] += r1[tid + s]; r2[tid] += r2[tid + s]; }
    __syncthreads();
  }
  if (tid == 0) { float m = r1[0] / (float)len; m_ = m; v_ = r2[0] / (float)len - m * m; }
  __syncthreads();
  float inv = rsqrtf(v_ + 1e-5f), mm = m_;
  for (long long i = tid; i < len; i += 256) {
    int c = gI * cg + (int)(i / S);
    float y = (x[base + i] - mm) * inv * gam[c] + bet[c];
    if (dosw) y = y / (1.f + expf(-y));
    x[base + i] = y;
  }
}

// ---------------- Farthest point sampling, coords: [B,3,Np], one block per b -----------
__global__ void k_fps(const float* __restrict__ coords, int* __restrict__ idx, int Np, int M) {
  __shared__ float mind[2048];
  __shared__ float rv[256];
  __shared__ int ri[256];
  __shared__ int sLast;
  int b = blockIdx.x, tid = threadIdx.x;
  const float* cb = coords + (size_t)b * 3 * Np;
  for (int n = tid; n < Np; n += 256) mind[n] = 1e10f;
  if (tid == 0) sLast = 0;
  __syncthreads();
  for (int j = 0; j < M; j++) {
    int last = sLast;
    if (tid == 0) idx[(size_t)b * M + j] = last;
    float cx = cb[last], cy = cb[Np + last], cz = cb[2 * Np + last];
    float bv = -1.f; int bi = 0;
    for (int n = tid; n < Np; n += 256) {
      float dx = cb[n] - cx, dy = cb[Np + n] - cy, dz = cb[2 * Np + n] - cz;
      float d = dx * dx + dy * dy + dz * dz;
      float mv = fminf(mind[n], d); mind[n] = mv;
      if (mv > bv) { bv = mv; bi = n; }
    }
    rv[tid] = bv; ri[tid] = bi; __syncthreads();
    for (int s = 128; s > 0; s >>= 1) {
      if (tid < s && rv[tid + s] > rv[tid]) { rv[tid] = rv[tid + s]; ri[tid] = ri[tid + s]; }
      __syncthreads();
    }
    if (tid == 0) sLast = ri[0];
    __syncthreads();
  }
}

__global__ void k_gather3(const float* __restrict__ coords, const int* __restrict__ idx,
                          float* __restrict__ ctr, int Np, int M, int B_) {
  int t = blockIdx.x * 256 + threadIdx.x;
  if (t >= B_ * 3 * M) return;
  int m = t % M; int c = (t / M) % 3; int b = t / (3 * M);
  ctr[t] = coords[((size_t)b * 3 + c) * Np + idx[(size_t)b * M + m]];
}

__global__ void k_ballquery(const float* __restrict__ coords, const float* __restrict__ ctr,
                            int* __restrict__ nidx, int B_, int Np, int M, int K, float r2) {
  int t = blockIdx.x * 256 + threadIdx.x;
  if (t >= B_ * M) return;
  int b = t / M, m = t % M;
  const float* cb = coords + (size_t)b * 3 * Np;
  float cx = ctr[((size_t)b * 3 + 0) * M + m];
  float cy = ctr[((size_t)b * 3 + 1) * M + m];
  float cz = ctr[((size_t)b * 3 + 2) * M + m];
  int* out = nidx + ((size_t)b * M + m) * K;
  int cnt = 0, first = -1;
  for (int n = 0; n < Np; n++) {
    float dx = cb[n] - cx, dy = cb[Np + n] - cy, dz = cb[2 * Np + n] - cz;
    if (dx * dx + dy * dy + dz * dz < r2) {
      if (first < 0) first = n;
      out[cnt++] = n;
      if (cnt >= K) break;
    }
  }
  if (first < 0) { for (int k = 0; k < K; k++) out[k] = 0; }
  else { for (int k = cnt; k < K; k++) out[k] = first; }
}

__global__ void k_group(const float* __restrict__ coords, const float* __restrict__ f,
                        const float* __restrict__ ctr, const int* __restrict__ nidx,
                        float* __restrict__ nfeat, int C, int Np, int M, int K, int B_) {
  int t = blockIdx.x * 256 + threadIdx.x;
  if (t >= B_ * M * K) return;
  int k = t % K; int m = (t / K) % M; int b = t / (K * M);
  int n = nidx[t];
  int Ct = 3 + C;
  for (int c = 0; c < 3; c++)
    nfeat[(((size_t)b * Ct + c) * M + m) * K + k] =
        coords[((size_t)b * 3 + c) * Np + n] - ctr[((size_t)b * 3 + c) * M + m];
  for (int c = 0; c < C; c++)
    nfeat[(((size_t)b * Ct + 3 + c) * M + m) * K + k] = f[((size_t)b * C + c) * Np + n];
}

__global__ void k_maxpool(const float* __restrict__ fm, float* __restrict__ out,
                          int C, int M, int K, int B_) {
  int t = blockIdx.x * 256 + threadIdx.x;
  if (t >= B_ * C * M) return;
  int m = t % M; int c = (t / M) % C; int b = t / (M * C);
  const float* p = fm + ((size_t)b * C + c) * (size_t)M * K + (size_t)m * K;
  float mx = p[0];
  for (int k = 1; k < K; k++) mx = fmaxf(mx, p[k]);
  out[((size_t)b * C + c) * M + m] = mx;
}

// 3-NN inverse-distance interpolation
__global__ void k_interp(const float* __restrict__ pts, const float* __restrict__ ctr,
                         const float* __restrict__ cf, float* __restrict__ out,
                         int B_, int Nout, int M, int Cc) {
  int t = blockIdx.x * 256 + threadIdx.x;
  if (t >= B_ * Nout) return;
  int b = t / Nout, n = t % Nout;
  float px = pts[((size_t)b * 3 + 0) * Nout + n];
  float py = pts[((size_t)b * 3 + 1) * Nout + n];
  float pz = pts[((size_t)b * 3 + 2) * Nout + n];
  float d0 = 1e30f, d1 = 1e30f, d2v = 1e30f; int i0 = 0, i1 = 0, i2 = 0;
  for (int m = 0; m < M; m++) {
    float dx = ctr[((size_t)b * 3 + 0) * M + m] - px;
    float dy = ctr[((size_t)b * 3 + 1) * M + m] - py;
    float dz = ctr[((size_t)b * 3 + 2) * M + m] - pz;
    float d = dx * dx + dy * dy + dz * dz;
    if (d < d0) { d2v = d1; i2 = i1; d1 = d0; i1 = i0; d0 = d; i0 = m; }
    else if (d < d1) { d2v = d1; i2 = i1; d1 = d; i1 = m; }
    else if (d < d2v) { d2v = d; i2 = m; }
  }
  float w0 = 1.f / (d0 + 1e-8f), w1 = 1.f / (d1 + 1e-8f), w2 = 1.f / (d2v + 1e-8f);
  float ws = w0 + w1 + w2; w0 /= ws; w1 /= ws; w2 /= ws;
  for (int c = 0; c < Cc; c++) {
    const float* row = cf + ((size_t)b * Cc + c) * M;
    out[((size_t)b * Cc + c) * Nout + n] = w0 * row[i0] + w1 * row[i1] + w2 * row[i2];
  }
}

// ---------------- voxelization ----------------
__global__ void k_vox_meta(const float* __restrict__ coords, float* __restrict__ nc,
                           int* __restrict__ flat, int Np, int r) {
  __shared__ float rx[256], ry[256], rz[256];
  __shared__ float mx_, my_, mz_, mn_;
  int b = blockIdx.x, tid = threadIdx.x;
  const float* cb = coords + (size_t)b * 3 * Np;
  float sx = 0, sy = 0, sz = 0;
  for (int n = tid; n < Np; n += 256) { sx += cb[n]; sy += cb[Np + n]; sz += cb[2 * Np + n]; }
  rx[tid] = sx; ry[tid] = sy; rz[tid] = sz; __syncthreads();
  for (int s = 128; s > 0; s >>= 1) {
    if (tid < s) { rx[tid] += rx[tid + s]; ry[tid] += ry[tid + s]; rz[tid] += rz[tid + s]; }
    __syncthreads();
  }
  if (tid == 0) { mx_ = rx[0] / Np; my_ = ry[0] / Np; mz_ = rz[0] / Np; }
  __syncthreads();
  float mnorm = 0.f;
  for (int n = tid; n < Np; n += 256) {
    float dx = cb[n] - mx_, dy = cb[Np + n] - my_, dz = cb[2 * Np + n] - mz_;
    mnorm = fmaxf(mnorm, sqrtf(dx * dx + dy * dy + dz * dz));
  }
  rx[tid] = mnorm; __syncthreads();
  for (int s = 128; s > 0; s >>= 1) {
    if (tid < s) rx[tid] = fmaxf(rx[tid], rx[tid + s]);
    __syncthreads();
  }
  if (tid == 0) mn_ = rx[0];
  __syncthreads();
  float inv = 1.f / (mn_ * 2.f);
  float* ncb = nc + (size_t)b * 3 * Np;
  float rm1 = (float)(r - 1);
  for (int n = tid; n < Np; n += 256) {
    float vx = ((cb[n] - mx_) * inv + 0.5f) * r;
    float vy = ((cb[Np + n] - my_) * inv + 0.5f) * r;
    float vz = ((cb[2 * Np + n] - mz_) * inv + 0.5f) * r;
    vx = fminf(fmaxf(vx, 0.f), rm1);
    vy = fminf(fmaxf(vy, 0.f), rm1);
    vz = fminf(fmaxf(vz, 0.f), rm1);
    ncb[n] = vx; ncb[Np + n] = vy; ncb[2 * Np + n] = vz;
    int ix = (int)rintf(vx), iy = (int)rintf(vy), iz = (int)rintf(vz);
    flat[(size_t)b * Np + n] = (ix * r + iy) * r + iz;
  }
}

__global__ void k_vox_scatter(const float* __restrict__ f, const int* __restrict__ flat,
                              float* __restrict__ grid, float* __restrict__ cnt,
                              int C, int Np, int r3, int B_) {
  int t = blockIdx.x * 256 + threadIdx.x;
  if (t >= B_ * Np) return;
  int b = t / Np, n = t % Np;
  int v = flat[t];
  atomicAdd(&cnt[(size_t)b * r3 + v], 1.f);
  for (int c = 0; c < C; c++)
    atomicAdd(&grid[((size_t)b * C + c) * r3 + v], f[((size_t)b * C + c) * Np + n]);
}

__global__ void k_vox_div(float* __restrict__ grid, const float* __restrict__ cnt,
                          int C, int r3, int B_) {
  int t = blockIdx.x * 256 + threadIdx.x;
  if (t >= B_ * C * r3) return;
  int v = t % r3; int b = t / (C * r3);
  grid[t] /= fmaxf(cnt[(size_t)b * r3 + v], 1.f);
}

__global__ void k_chanmean(const float* __restrict__ h, float* __restrict__ s,
                           int C, int r3, int B_) {
  int t = blockIdx.x * 256 + threadIdx.x;
  if (t >= B_ * C) return;
  const float* p = h + (size_t)t * r3;
  float acc = 0.f;
  for (int v = 0; v < r3; v++) acc += p[v];
  s[t] = acc / (float)r3;
}

// small dense: out[b,o] = act(in[b,:]@W[o,:] + bias[o]); act: 0 none, 1 lrelu0.1, 2 relu, 3 sigmoid
__global__ void k_dense(const float* __restrict__ in, const float* __restrict__ W,
                        const float* __restrict__ bias, float* __restrict__ out,
                        int I, int O, int B_, int act) {
  int t = blockIdx.x * 256 + threadIdx.x;
  if (t >= B_ * O) return;
  int b = t / O, o = t % O;
  float acc = bias ? bias[o] : 0.f;
  for (int i = 0; i < I; i++) acc += in[(size_t)b * I + i] * W[(size_t)o * I + i];
  if (act == 1) acc = acc >= 0.f ? acc : 0.1f * acc;
  else if (act == 2) acc = fmaxf(acc, 0.f);
  else if (act == 3) acc = 1.f / (1.f + expf(-acc));
  out[t] = acc;
}

__global__ void k_scale(float* __restrict__ h, const float* __restrict__ s, int r3, int n) {
  int t = blockIdx.x * 256 + threadIdx.x;
  if (t < n) h[t] *= s[t / r3];
}

__global__ void k_devox(const float* __restrict__ grid, const float* __restrict__ nc,
                        float* __restrict__ out, int B_, int C, int Np, int r) {
  int t = blockIdx.x * 256 + threadIdx.x;
  if (t >= B_ * Np) return;
  int b = t / Np, n = t % Np;
  int r3 = r * r * r;
  float fx = nc[((size_t)b * 3 + 0) * Np + n];
  float fy = nc[((size_t)b * 3 + 1) * Np + n];
  float fz = nc[((size_t)b * 3 + 2) * Np + n];
  int x0 = (int)floorf(fx), y0 = (int)floorf(fy), z0 = (int)floorf(fz);
  float ax = fx - x0, ay = fy - y0, az = fz - z0;
  int x1 = min(x0 + 1, r - 1), y1 = min(y0 + 1, r - 1), z1 = min(z0 + 1, r - 1);
  int ci[8]; float cw[8];
  int xs[2] = {x0, x1}, ys[2] = {y0, y1}, zs[2] = {z0, z1};
  float wx[2] = {1.f - ax, ax}, wy[2] = {1.f - ay, ay}, wz[2] = {1.f - az, az};
  int q = 0;
  for (int dx = 0; dx < 2; dx++)
    for (int dy = 0; dy < 2; dy++)
      for (int dz = 0; dz < 2; dz++) {
        ci[q] = (xs[dx] * r + ys[dy]) * r + zs[dz];
        cw[q] = wx[dx] * wy[dy] * wz[dz];
        q++;
      }
  for (int c = 0; c < C; c++) {
    const float* g = grid + ((size_t)b * C + c) * r3;
    float acc = 0.f;
#pragma unroll
    for (int k = 0; k < 8; k++) acc += cw[k] * g[ci[k]];
    out[((size_t)b * C + c) * Np + n] = acc;
  }
}

__global__ void k_timestep(const int* __restrict__ t, float* __restrict__ emb, int B_) {
  int i = blockIdx.x * 256 + threadIdx.x;
  if (i >= B_ * 64) return;
  int b = i / 64, j = i % 64;
  int h = j % 32;
  float fr = expf(-logf(10000.f) * (float)h / 31.f);
  float a = (float)t[b] * fr;
  emb[i] = (j >= 32) ? cosf(a) : sinf(a);
}

// ---------------- host orchestration ----------------
struct MLPL { const float *w, *b, *g, *be; int ci, co; };
struct PVP {
  const float *w1, *b1, *g1, *be1, *w2, *b2, *g2, *be2, *se1, *se2;
  MLPL point; int ci, co, red, r;
};

static inline unsigned grd(size_t n) { return (unsigned)((n + 255) / 256); }
static inline int rnd32(int v) { return (v + 31) & ~31; }
static inline int rnd16(int v) { return (v + 15) & ~15; }

extern "C" void kernel_launch(void* const* d_in, const int* in_sizes, int n_in,
                              void* d_out, int out_size, void* d_ws, size_t ws_size,
                              hipStream_t stream) {
  (void)in_sizes; (void)n_in; (void)out_size; (void)ws_size;
  const int Bc = 8, Nfull = 2048, Vv = 3, Kq = 32;
  const float* inputs = (const float*)d_in[0];
  const int* tptr = (const int*)d_in[1];

  // ---- parameter walk (setup_inputs insertion order) ----
  int pi = 2;
  auto F = [&]() -> const float* { return (const float*)d_in[pi++]; };
  auto readL = [&](int ci, int co) -> MLPL {
    MLPL L; L.w = F(); L.b = F(); L.g = F(); L.be = F(); L.ci = ci; L.co = co; return L;
  };
  auto readPv = [&](int ci, int co, int r) -> PVP {
    PVP p;
    p.w1 = F(); p.b1 = F(); p.g1 = F(); p.be1 = F();
    p.w2 = F(); p.b2 = F(); p.g2 = F(); p.be2 = F();
    p.se1 = F(); p.se2 = F();
    p.point.w = F(); p.point.b = F(); p.point.g = F(); p.point.be = F();
    p.point.ci = ci; p.point.co = co;
    p.ci = ci; p.co = co; p.red = (co / 8 > 0) ? co / 8 : 1; p.r = r;
    return p;
  };
  const float* ew1 = F(); const float* eb1 = F();
  const float* ew2 = F(); const float* eb2 = F();

  const int SA_M[4] = {1024, 256, 64, 16};
  const float SA_RAD[4] = {0.1f, 0.2f, 0.4f, 0.8f};
  const int saNConv[4] = {2, 1, 1, 0};
  const int saConvSpec[4][2][3] = {
      {{3, 32, 32}, {32, 32, 32}}, {{128, 64, 16}, {0, 0, 0}},
      {{192, 128, 8}, {0, 0, 0}}, {{0, 0, 0}, {0, 0, 0}}};
  const int saMlpIn[4] = {35, 67, 131, 323};
  const int saMlpN[4] = {2, 2, 2, 3};
  const int saMlpCo[4][3] = {{32, 64, 0}, {64, 128, 0}, {128, 256, 0}, {256, 256, 512}};
  const int fpMlpIn[4] = {832, 448, 384, 192};
  const int fpMlpN[4] = {2, 2, 2, 3};
  const int fpMlpCo[4][3] = {{256, 256, 0}, {256, 256, 0}, {256, 128, 0}, {128, 128, 64}};
  const int fpNConv[4] = {3, 3, 2, 2};
  const int fpConvSpec[4][3][3] = {
      {{256, 256, 8}, {256, 256, 8}, {256, 256, 8}},
      {{256, 256, 8}, {256, 256, 8}, {256, 256, 8}},
      {{128, 128, 16}, {128, 128, 16}, {0, 0, 0}},
      {{64, 64, 32}, {64, 64, 32}, {0, 0, 0}}};

  PVP saPv[4][2]; MLPL saL[4][3];
  for (int s = 0; s < 4; s++) {
    for (int cv = 0; cv < saNConv[s]; cv++)
      saPv[s][cv] = readPv(saConvSpec[s][cv][0], saConvSpec[s][cv][1], saConvSpec[s][cv][2]);
    int ci = saMlpIn[s];
    for (int l = 0; l < saMlpN[s]; l++) { saL[s][l] = readL(ci, saMlpCo[s][l]); ci = saMlpCo[s][l]; }
  }
  PVP fpPv[4][3]; MLPL fpL[4][3];
  for (int s = 0; s < 4; s++) {
    int ci = fpMlpIn[s];
    for (int l = 0; l < fpMlpN[s]; l++) { fpL[s][l] = readL(ci, fpMlpCo[s][l]); ci = fpMlpCo[s][l]; }
    for (int cv = 0; cv < fpNConv[s]; cv++)
      fpPv[s][cv] = readPv(fpConvSpec[s][cv][0], fpConvSpec[s][cv][1], fpConvSpec[s][cv][2]);
  }
  MLPL clsL = readL(64, 128);
  const float* clsW = F();
  const float* clsB = F();

  // ---- bump allocator on d_ws (stream ordering makes scoped reuse safe) ----
  size_t wsOff = 0;
  auto alloc = [&](size_t bytes) -> void* {
    wsOff = (wsOff + 255) & ~(size_t)255;
    void* p = (char*)d_ws + wsOff;
    wsOff += bytes;
    return p;
  };
  auto fill0 = [&](float* p, size_t n) {
    k_fill<<<grd(n), 256, 0, stream>>>(p, 0.f, (int)n);
  };
  auto gemm = [&](const float* W, const float* X, const float* bias, float* Y,
                  int O, int K, int Nn) {
    size_t mk = wsOff;
    int Kp = rnd32(K), Op = rnd16(O);
    int gx = (Nn + 63) / 64, Nnp = gx * 64;
    _Float16* Wh = (_Float16*)alloc((size_t)Op * Kp * 2);
    _Float16* Xt = (_Float16*)alloc((size_t)Bc * Nnp * Kp * 2);
    k_wpad<<<grd((size_t)Op * Kp), 256, 0, stream>>>(Wh, W, O, K, Op, Kp);
    k_xtr<<<grd((size_t)Bc * Nnp * Kp), 256, 0, stream>>>(Xt, X, K, Kp, Nn, Nnp, Bc);
    k_gemm_wmma<<<dim3(gx, Op / 16, Bc), 32, 0, stream>>>(Wh, Xt, bias, Y, O, Kp, Nn, Nnp);
    wsOff = mk;
  };
  auto gn = [&](float* Y, const float* gamma, const float* beta, int C, int S) {
    k_groupnorm<<<dim3(8, Bc), 256, 0, stream>>>(Y, gamma, beta, C, S, 1);
  };
  auto run_mlp = [&](const MLPL* Ls, int nL, const float* X, int S, float* finalOut) -> float* {
    const float* cur = X;
    float* Y = nullptr;
    for (int l = 0; l < nL; l++) {
      const MLPL& L = Ls[l];
      Y = (l == nL - 1 && finalOut) ? finalOut : (float*)alloc((size_t)Bc * L.co * S * 4);
      gemm(L.w, cur, L.b, Y, L.co, L.ci, S);
      gn(Y, L.g, L.be, L.co, S);
      cur = Y;
    }
    return Y;
  };
  auto conv3 = [&](const float* W, const float* bias, const float* G, int Cin,
                   float* Y, int O, int r) {
    size_t mk = wsOff;
    int r3 = r * r * r, Cp = rnd32(Cin), Op = rnd16(O);
    _Float16* Wt = (_Float16*)alloc((size_t)Op * 27 * Cp * 2);
    _Float16* Gt = (_Float16*)alloc((size_t)Bc * r3 * Cp * 2);
    k_wconv<<<grd((size_t)Op * 27 * Cp), 256, 0, stream>>>(Wt, W, O, Cin, Op, Cp);
    k_gtr<<<grd((size_t)Bc * r3 * Cp), 256, 0, stream>>>(Gt, G, Cin, Cp, r3, Bc);
    k_conv3d_wmma<<<dim3(r3 / 64, Op / 16, Bc), 32, 0, stream>>>(Wt, Gt, bias, Y, O, Cp, r);
    wsOff = mk;
  };
  auto pvconv = [&](const PVP& p, const float* f, const float* coords, int Np, float* out) {
    size_t m0 = wsOff;
    int r = p.r, r3 = r * r * r;
    float* grid = (float*)alloc((size_t)Bc * p.ci * r3 * 4);
    float* cnt = (float*)alloc((size_t)Bc * r3 * 4);
    float* nc = (float*)alloc((size_t)Bc * 3 * Np * 4);
    int* flat = (int*)alloc((size_t)Bc * Np * 4);
    fill0(grid, (size_t)Bc * p.ci * r3);
    fill0(cnt, (size_t)Bc * r3);
    k_vox_meta<<<Bc, 256, 0, stream>>>(coords, nc, flat, Np, r);
    k_vox_scatter<<<grd((size_t)Bc * Np), 256, 0, stream>>>(f, flat, grid, cnt, p.ci, Np, r3, Bc);
    k_vox_div<<<grd((size_t)Bc * p.ci * r3), 256, 0, stream>>>(grid, cnt, p.ci, r3, Bc);
    float* h1 = (float*)alloc((size_t)Bc * p.co * r3 * 4);
    conv3(p.w1, p.b1, grid, p.ci, h1, p.co, r);
    gn(h1, p.g1, p.be1, p.co, r3);
    float* h2 = (float*)alloc((size_t)Bc * p.co * r3 * 4);
    conv3(p.w2, p.b2, h1, p.co, h2, p.co, r);
    gn(h2, p.g2, p.be2, p.co, r3);
    float* s = (float*)alloc((size_t)Bc * p.co * 4);
    float* s1 = (float*)alloc((size_t)Bc * p.red * 4);
    float* s2 = (float*)alloc((size_t)Bc * p.co * 4);
    k_chanmean<<<grd((size_t)Bc * p.co), 256, 0, stream>>>(h2, s, p.co, r3, Bc);
    k_dense<<<grd((size_t)Bc * p.red), 256, 0, stream>>>(s, p.se1, nullptr, s1, p.co, p.red, Bc, 2);
    k_dense<<<grd((size_t)Bc * p.co), 256, 0, stream>>>(s1, p.se2, nullptr, s2, p.red, p.co, Bc, 3);
    k_scale<<<grd((size_t)Bc * p.co * r3), 256, 0, stream>>>(h2, s2, r3, Bc * p.co * r3);
    k_devox<<<grd((size_t)Bc * Np), 256, 0, stream>>>(h2, nc, out, Bc, p.co, Np, r);
    float* pt = run_mlp(&p.point, 1, f, Np, nullptr);
    k_add<<<grd((size_t)Bc * p.co * Np), 256, 0, stream>>>(out, pt, Bc * p.co * Np);
    wsOff = m0;
  };

  // ---- time embedding (constant across network; broadcast where concatenated) ----
  float* embBuf = (float*)alloc((size_t)Bc * 64 * 4);
  float* e1 = (float*)alloc((size_t)Bc * 64 * 4);
  float* te = (float*)alloc((size_t)Bc * 64 * 4);
  k_timestep<<<grd((size_t)Bc * 64), 256, 0, stream>>>(tptr, embBuf, Bc);
  k_dense<<<grd((size_t)Bc * 64), 256, 0, stream>>>(embBuf, ew1, eb1, e1, 64, 64, Bc, 1);
  k_dense<<<grd((size_t)Bc * 64), 256, 0, stream>>>(e1, ew2, eb2, te, 64, 64, Bc, 0);

  size_t viewMark = wsOff;
  for (int j = 0; j < Vv; j++) {
    wsOff = viewMark;
    const float* x = inputs + (size_t)j * Bc * 3 * Nfull;
    const float* coordsL[4]; int NpL[4];
    const float* featsL[4]; int featsC[4];
    const float* coords = x; int Np = Nfull;
    float* features = nullptr; int C = 0;

    // ---- set abstraction trunk ----
    for (int s = 0; s < 4; s++) {
      coordsL[s] = coords; NpL[s] = Np; featsL[s] = features; featsC[s] = C;
      int M = SA_M[s];
      int coSA = saMlpCo[s][saMlpN[s] - 1];
      float* featOut = (float*)alloc((size_t)Bc * coSA * M * 4);
      float* ctrOut = (float*)alloc((size_t)Bc * 3 * M * 4);
      size_t m0 = wsOff;
      const float* f; int fc;
      if (s == 0) { f = x; fc = 3; }
      else {
        fc = C + 64;
        float* fb = (float*)alloc((size_t)Bc * fc * Np * 4);
        k_copy_ch<<<grd((size_t)Bc * C * Np), 256, 0, stream>>>(fb, features, 0, fc, C, Np, Bc);
        k_bcast<<<grd((size_t)Bc * 64 * Np), 256, 0, stream>>>(fb, te, C, fc, Np, Bc);
        f = fb;
      }
      for (int cv = 0; cv < saNConv[s]; cv++) {
        const PVP& p = saPv[s][cv];
        float* o = (float*)alloc((size_t)Bc * p.co * Np * 4);
        pvconv(p, f, coords, Np, o);
        f = o; fc = p.co;
      }
      int* idx = (int*)alloc((size_t)Bc * M * 4);
      k_fps<<<Bc, 256, 0, stream>>>(coords, idx, Np, M);
      k_gather3<<<grd((size_t)Bc * 3 * M), 256, 0, stream>>>(coords, idx, ctrOut, Np, M, Bc);
      int* nid = (int*)alloc((size_t)Bc * M * Kq * 4);
      k_ballquery<<<grd((size_t)Bc * M), 256, 0, stream>>>(
          coords, ctrOut, nid, Bc, Np, M, Kq, SA_RAD[s] * SA_RAD[s]);
      int gc = 3 + fc;
      float* nf = (float*)alloc((size_t)Bc * gc * M * Kq * 4);
      k_group<<<grd((size_t)Bc * M * Kq), 256, 0, stream>>>(
          coords, f, ctrOut, nid, nf, fc, Np, M, Kq, Bc);
      float* fm = run_mlp(saL[s], saMlpN[s], nf, M * Kq, nullptr);
      k_maxpool<<<grd((size_t)Bc * coSA * M), 256, 0, stream>>>(fm, featOut, coSA, M, Kq, Bc);
      wsOff = m0;
      features = featOut; C = coSA; coords = ctrOut; Np = M;
    }

    // ---- feature propagation ----
    for (int fi = 0; fi < 4; fi++) {
      const float* pts = coordsL[3 - fi]; int Nout = NpL[3 - fi];
      int coF = fpMlpCo[fi][fpMlpN[fi] - 1];
      int finalC = fpNConv[fi] ? fpConvSpec[fi][fpNConv[fi] - 1][1] : coF;
      float* stageOut = (float*)alloc((size_t)Bc * finalC * Nout * 4);
      size_t m0 = wsOff;
      int Cc = C + 64;
      float* cf = (float*)alloc((size_t)Bc * Cc * Np * 4);
      k_copy_ch<<<grd((size_t)Bc * C * Np), 256, 0, stream>>>(cf, features, 0, Cc, C, Np, Bc);
      k_bcast<<<grd((size_t)Bc * 64 * Np), 256, 0, stream>>>(cf, te, C, Cc, Np, Bc);
      float* itp = (float*)alloc((size_t)Bc * Cc * Nout * 4);
      k_interp<<<grd((size_t)Bc * Nout), 256, 0, stream>>>(pts, coords, cf, itp, Bc, Nout, Np, Cc);
      int Cs = featsC[3 - fi];
      int ciM = Cc + Cs;
      float* cat = (float*)alloc((size_t)Bc * ciM * Nout * 4);
      k_copy_ch<<<grd((size_t)Bc * Cc * Nout), 256, 0, stream>>>(cat, itp, 0, ciM, Cc, Nout, Bc);
      if (Cs > 0)
        k_copy_ch<<<grd((size_t)Bc * Cs * Nout), 256, 0, stream>>>(
            cat, featsL[3 - fi], Cc, ciM, Cs, Nout, Bc);
      const float* cur = run_mlp(fpL[fi], fpMlpN[fi], cat, Nout,
                                 (fpNConv[fi] == 0) ? stageOut : nullptr);
      for (int cv = 0; cv < fpNConv[fi]; cv++) {
        float* o = (cv == fpNConv[fi] - 1)
                       ? stageOut
                       : (float*)alloc((size_t)Bc * fpPv[fi][cv].co * Nout * 4);
        pvconv(fpPv[fi][cv], cur, pts, Nout, o);
        cur = o;
      }
      wsOff = m0;
      features = stageOut; C = finalC; coords = pts; Np = Nout;
    }

    // ---- classifier head ----
    {
      size_t m0 = wsOff;
      float* h = run_mlp(&clsL, 1, features, Nfull, nullptr);
      gemm(clsW, h, clsB, (float*)d_out + (size_t)j * Bc * 3 * Nfull, 3, 128, Nfull);
      wsOff = m0;
    }
  }
}
